// ChebyKANLayer_64355789963436
// MI455X (gfx1250) — matrix operand order repaired and checked by reading, not compiled, not run on hardware
//
#include <hip/hip_runtime.h>
#include <hip/hip_bf16.h>
#include <stdint.h>

// ---------------------------------------------------------------------------
// ChebyKAN layer on MI455X (gfx1250):
//   h = tanh(layernorm(x))                          [B=4096, I=1024] bf16
//   y[b,o] = bias0[o] + sum_{d=1..4} T_d(h[b,:]) @ C[:,o,d]
//   GEMM: M=4096, N=1024, K=4096 via v_wmma_f32_16x16x32_bf16,
//         double-buffered LDS fed by global_load_async_to_lds_b128 (ASYNCcnt)
//   out = silu(y)  (f32)
// ---------------------------------------------------------------------------

#define BDIM   4096
#define IDIM   1024
#define ODIM   1024
#define NDEG   5                  // DEGREE+1
#define KGEMM  (IDIM * 4)         // 4096 (degrees 1..4)

#define BM 128
#define BN 128
#define BK 32
#define NKSEG (IDIM / BK)         // 32 K-iterations per degree segment
#define LDS_STRIDE 40             // BK + 8 pad (ushort) -> 80B rows, conflict-free b128

typedef __attribute__((ext_vector_type(16))) __bf16 v16bf;
typedef __attribute__((ext_vector_type(8)))  __bf16 v8bf;
typedef __attribute__((ext_vector_type(8)))  float  v8f;

__device__ __forceinline__ unsigned short f2bf(float f) {
    unsigned u = __builtin_bit_cast(unsigned, f);
    unsigned r = (u + 0x7FFFu + ((u >> 16) & 1u)) >> 16;   // RNE
    return (unsigned short)r;
}
__device__ __forceinline__ float bf2f(unsigned hs) {
    return __builtin_bit_cast(float, hs << 16);
}
__device__ __forceinline__ v8bf ld_v8bf(const unsigned short* p) {
    uint4 u = *(const uint4*)p;                             // 16B aligned
    return __builtin_bit_cast(v8bf, u);
}

// CDNA5 async DMA: 16B global -> LDS, tracked by ASYNCcnt (no VGPR round trip).
// %0 = 32-bit LDS byte address (low 32 bits of a flat shared pointer),
// %1 = 64-bit global address.
__device__ __forceinline__ void async_g2l_b128(unsigned lds_addr, const void* gptr) {
    asm volatile("global_load_async_to_lds_b128 %0, %1, off"
                 :: "v"(lds_addr), "v"(gptr)
                 : "memory");
}
__device__ __forceinline__ void wait_async0() {
    asm volatile("s_wait_asynccnt 0" ::: "memory");
}
__device__ __forceinline__ unsigned lds_addr_of(const void* p) {
    return (unsigned)(uintptr_t)p;
}

// ---------------------------------------------------------------------------
// Kernel 0: repack coeffs [I,O,5] f32 -> Wt [O, K=4*I] bf16 (degrees 1..4),
//           plus deterministic per-i-tile partial sums of the d=0 plane.
// ---------------------------------------------------------------------------
__global__ void __launch_bounds__(256)
cheby_repack_kernel(const float* __restrict__ coeffs, unsigned short* __restrict__ wt,
                    float* __restrict__ pbias /* [I/32][O] */) {
    __shared__ float tile[32][32][NDEG + 1];
    const int i0 = blockIdx.x * 32;
    const int o0 = blockIdx.y * 32;
    const int t  = threadIdx.x;

    for (int f = t; f < 32 * 32 * NDEG; f += 256) {
        int ii  = f / (32 * NDEG);
        int rem = f - ii * (32 * NDEG);
        int oo  = rem / NDEG;
        int d   = rem - oo * NDEG;
        tile[ii][oo][d] = coeffs[((size_t)(i0 + ii) * ODIM + (o0 + oo)) * NDEG + d];
    }
    __syncthreads();
    for (int g = t; g < 32 * 32 * 4; g += 256) {
        int oo  = g / (4 * 32);
        int rem = g - oo * (4 * 32);
        int d   = rem / 32;              // 0..3 -> degree d+1
        int ii  = rem - d * 32;
        wt[(size_t)(o0 + oo) * KGEMM + d * IDIM + (i0 + ii)] =
            f2bf(tile[ii][oo][d + 1]);
    }
    if (t < 32) {                        // d=0 partial column sums, deterministic
        float s = 0.0f;
        #pragma unroll
        for (int ii = 0; ii < 32; ++ii) s += tile[ii][t][0];
        pbias[(size_t)blockIdx.x * ODIM + (o0 + t)] = s;
    }
}

__global__ void __launch_bounds__(256)
cheby_bias_reduce(const float* __restrict__ pbias, float* __restrict__ bias0) {
    const int o = blockIdx.x * 256 + threadIdx.x;
    float s = 0.0f;
    #pragma unroll
    for (int it = 0; it < IDIM / 32; ++it) s += pbias[(size_t)it * ODIM + o];
    bias0[o] = s;
}

// ---------------------------------------------------------------------------
// Kernel 1: h = tanh(layernorm(x)) -> bf16  (one 256-thread block per row)
// ---------------------------------------------------------------------------
__global__ void __launch_bounds__(256)
ln_tanh_kernel(const float* __restrict__ x, const float* __restrict__ w,
               const float* __restrict__ b, unsigned short* __restrict__ hout) {
    __shared__ float rs[256];
    __shared__ float rq[256];
    const int row = blockIdx.x;
    const int tid = threadIdx.x;

    float4 v = ((const float4*)(x + (size_t)row * IDIM))[tid];
    rs[tid] = v.x + v.y + v.z + v.w;
    rq[tid] = v.x * v.x + v.y * v.y + v.z * v.z + v.w * v.w;
    __syncthreads();
    for (int st = 128; st > 0; st >>= 1) {
        if (tid < st) { rs[tid] += rs[tid + st]; rq[tid] += rq[tid + st]; }
        __syncthreads();
    }
    const float mean = rs[0] * (1.0f / IDIM);
    const float var  = rq[0] * (1.0f / IDIM) - mean * mean;
    const float rsd  = rsqrtf(var + 1e-5f);

    float4 wv = ((const float4*)w)[tid];
    float4 bv = ((const float4*)b)[tid];
    float h0 = tanhf((v.x - mean) * rsd * wv.x + bv.x);
    float h1 = tanhf((v.y - mean) * rsd * wv.y + bv.y);
    float h2 = tanhf((v.z - mean) * rsd * wv.z + bv.z);
    float h3 = tanhf((v.w - mean) * rsd * wv.w + bv.w);

    uint2 packed;
    packed.x = (unsigned)f2bf(h0) | ((unsigned)f2bf(h1) << 16);
    packed.y = (unsigned)f2bf(h2) | ((unsigned)f2bf(h3) << 16);
    ((uint2*)(hout + (size_t)row * IDIM))[tid] = packed;
}

// ---------------------------------------------------------------------------
// Kernel 2: bf16 WMMA GEMM, compile-time-degree K segments, double-buffered
//           LDS fed by async global->LDS DMA, SiLU epilogue.
// ---------------------------------------------------------------------------
template <int D>
__device__ __forceinline__ void load_tile(
    const unsigned short* __restrict__ hrow,
    const unsigned short* __restrict__ wrow,
    unsigned short* sAbuf, unsigned short* sBbuf,
    int ldRow, int ldKc, int kt) {

    const int k0 = kt * BK + ldKc;

    // B tile: pure copy of Wt -> async DMA straight into LDS
    const unsigned boff = lds_addr_of(&sBbuf[ldRow * LDS_STRIDE + ldKc]);
    async_g2l_b128(boff,      wrow + k0);
    async_g2l_b128(boff + 16, wrow + k0 + 8);

    if constexpr (D == 1) {
        // T1 = h: pure copy -> async DMA
        const unsigned aoff = lds_addr_of(&sAbuf[ldRow * LDS_STRIDE + ldKc]);
        async_g2l_b128(aoff,      hrow + k0);
        async_g2l_b128(aoff + 16, hrow + k0 + 8);
    } else {
        // T_D needs compute: global -> regs -> transform -> ds_store
        uint4 ha = *(const uint4*)(hrow + k0);
        uint4 hc = *(const uint4*)(hrow + k0 + 8);
        unsigned hw[8] = {ha.x, ha.y, ha.z, ha.w, hc.x, hc.y, hc.z, hc.w};
        unsigned aw[8];
        #pragma unroll
        for (int j = 0; j < 8; ++j) {
            float x0 = bf2f(hw[j] & 0xFFFFu);
            float x1 = bf2f(hw[j] >> 16);
            float e0, e1;
            if constexpr (D == 2) {                     // 2h^2 - 1
                e0 = __builtin_fmaf(2.0f * x0, x0, -1.0f);
                e1 = __builtin_fmaf(2.0f * x1, x1, -1.0f);
            } else if constexpr (D == 3) {              // h(4h^2 - 3)
                e0 = x0 * __builtin_fmaf(4.0f * x0, x0, -3.0f);
                e1 = x1 * __builtin_fmaf(4.0f * x1, x1, -3.0f);
            } else {                                    // 8h^2(h^2-1) + 1
                float s0 = x0 * x0, s1 = x1 * x1;
                e0 = __builtin_fmaf(8.0f * s0, s0 - 1.0f, 1.0f);
                e1 = __builtin_fmaf(8.0f * s1, s1 - 1.0f, 1.0f);
            }
            aw[j] = (unsigned)f2bf(e0) | ((unsigned)f2bf(e1) << 16);
        }
        unsigned short* pa = &sAbuf[ldRow * LDS_STRIDE + ldKc];
        *(uint4*)(pa)     = make_uint4(aw[0], aw[1], aw[2], aw[3]);
        *(uint4*)(pa + 8) = make_uint4(aw[4], aw[5], aw[6], aw[7]);
    }
}

template <int D>
__device__ __forceinline__ void gemm_segment(
    const unsigned short* __restrict__ hrow,
    const unsigned short* __restrict__ wrow,
    unsigned short* sA0, unsigned short* sA1,
    unsigned short* sB0, unsigned short* sB1,
    int ldRow, int ldKc, int lane, int waveM, int waveN, v8f (&acc)[8]) {

    unsigned short* sAb[2] = {sA0, sA1};
    unsigned short* sBb[2] = {sB0, sB1};

    // prologue: fill buffer 0
    __syncthreads();                       // prior reads of buffers complete
    load_tile<D>(hrow, wrow, sAb[0], sBb[0], ldRow, ldKc, 0);
    wait_async0();
    __syncthreads();                       // tile 0 visible to all waves

    int p = 0;
    for (int kt = 0; kt < NKSEG; ++kt) {
        if (kt + 1 < NKSEG)                // overlap DMA of next tile with WMMA
            load_tile<D>(hrow, wrow, sAb[p ^ 1], sBb[p ^ 1], ldRow, ldKc, kt + 1);

        unsigned short* sAc = sAb[p];
        unsigned short* sBc = sBb[p];

        // A fragment (16x32 bf16): lane<16 -> K 0-7 & 16-23 ; lane>=16 -> +8
        v16bf afr[4];
        {
            const int ar = lane & 15;
            const int kb = (lane >> 4) * 8;
            #pragma unroll
            for (int m = 0; m < 4; ++m) {
                const unsigned short* ptr = &sAc[(waveM + m * 16 + ar) * LDS_STRIDE + kb];
                v8bf lo = ld_v8bf(ptr);
                v8bf hi = ld_v8bf(ptr + 16);
                v16bf a;
                #pragma unroll
                for (int j = 0; j < 8; ++j) { a[j] = lo[j]; a[8 + j] = hi[j]; }
                afr[m] = a;
            }
        }
        // B fragment (32x16 bf16): lane<16 -> K 0-7 & 16-23 ; lane>=16 -> +8
        v16bf bfr[2];
        {
            const int bc = lane & 15;
            const int kb = (lane >> 4) * 8;
            #pragma unroll
            for (int n = 0; n < 2; ++n) {
                const unsigned short* ptr = &sBc[(waveN + n * 16 + bc) * LDS_STRIDE + kb];
                v8bf lo = ld_v8bf(ptr);
                v8bf hi = ld_v8bf(ptr + 16);
                v16bf bvv;
                #pragma unroll
                for (int j = 0; j < 8; ++j) { bvv[j] = lo[j]; bvv[8 + j] = hi[j]; }
                bfr[n] = bvv;
            }
        }

        #pragma unroll
        for (int m = 0; m < 4; ++m)
            #pragma unroll
            for (int n = 0; n < 2; ++n)
                acc[m * 2 + n] = __builtin_amdgcn_wmma_f32_16x16x32_bf16(
                    false, afr[m], false, bfr[n], (short)0, acc[m * 2 + n],
                    false, false);

        if (kt + 1 < NKSEG) {
            wait_async0();                 // our DMA into shadow buffer done
            __syncthreads();               // everyone's writes visible; reads of p done
            p ^= 1;
        }
    }
}

__global__ void __launch_bounds__(256)
chebykan_wmma_gemm(const unsigned short* __restrict__ hbf,
                   const unsigned short* __restrict__ wt,
                   const float* __restrict__ bias0,
                   float* __restrict__ out) {
    __shared__ unsigned short sA[2][BM * LDS_STRIDE];   // 2 x 10 KB
    __shared__ unsigned short sB[2][BN * LDS_STRIDE];   // 2 x 10 KB

    const int t      = threadIdx.x;
    const int lane   = t & 31;
    const int wave   = t >> 5;
    const int waveM  = (wave >> 2) * 64;  // 2 waves along M
    const int waveN  = (wave & 3) * 32;   // 4 waves along N
    const int blockN = blockIdx.x * BN;
    const int blockM = blockIdx.y * BM;

    const int ldRow = t >> 1;             // 0..127
    const int ldKc  = (t & 1) * 16;       // 0 or 16

    v8f acc[8];
    const v8f vzero = (v8f)0.0f;
    #pragma unroll
    for (int i = 0; i < 8; ++i) acc[i] = vzero;

    const unsigned short* hrow = hbf + (size_t)(blockM + ldRow) * IDIM;
    const unsigned short* wrow = wt  + (size_t)(blockN + ldRow) * KGEMM;

    gemm_segment<1>(hrow, wrow + 0 * IDIM, sA[0], sA[1], sB[0], sB[1],
                    ldRow, ldKc, lane, waveM, waveN, acc);
    gemm_segment<2>(hrow, wrow + 1 * IDIM, sA[0], sA[1], sB[0], sB[1],
                    ldRow, ldKc, lane, waveM, waveN, acc);
    gemm_segment<3>(hrow, wrow + 2 * IDIM, sA[0], sA[1], sB[0], sB[1],
                    ldRow, ldKc, lane, waveM, waveN, acc);
    gemm_segment<4>(hrow, wrow + 3 * IDIM, sA[0], sA[1], sB[0], sB[1],
                    ldRow, ldKc, lane, waveM, waveN, acc);

    // epilogue: + bias0 (hoisted T0 term), SiLU, f32 store
    const int cr = (lane >> 4) * 8;
    const int cc = lane & 15;
    #pragma unroll
    for (int n = 0; n < 2; ++n) {
        const int col = blockN + waveN + n * 16 + cc;
        const float b0 = bias0[col];
        #pragma unroll
        for (int m = 0; m < 4; ++m) {
            v8f a = acc[m * 2 + n];
            #pragma unroll
            for (int j = 0; j < 8; ++j) {
                const int row = blockM + waveM + m * 16 + cr + j;
                float y = a[j] + b0;
                float s = y / (1.0f + __expf(-y));
                out[(size_t)row * ODIM + col] = s;
            }
        }
    }
}

// ---------------------------------------------------------------------------
extern "C" void kernel_launch(void* const* d_in, const int* in_sizes, int n_in,
                              void* d_out, int out_size, void* d_ws, size_t ws_size,
                              hipStream_t stream) {
    const float* x      = (const float*)d_in[0];
    const float* coeffs = (const float*)d_in[1];
    const float* lnw    = (const float*)d_in[2];
    const float* lnb    = (const float*)d_in[3];
    float* out          = (float*)d_out;

    // workspace: hbf 8MB | wt 8MB | pbias 128KB | bias0 4KB
    unsigned short* hbf = (unsigned short*)d_ws;
    unsigned short* wt  = hbf + (size_t)BDIM * IDIM;
    float* pbias        = (float*)(wt + (size_t)ODIM * KGEMM);
    float* bias0        = pbias + (size_t)(IDIM / 32) * ODIM;

    cheby_repack_kernel<<<dim3(IDIM / 32, ODIM / 32), 256, 0, stream>>>(coeffs, wt, pbias);
    cheby_bias_reduce<<<ODIM / 256, 256, 0, stream>>>(pbias, bias0);
    ln_tanh_kernel<<<BDIM, 256, 0, stream>>>(x, lnw, lnb, hbf);
    chebykan_wmma_gemm<<<dim3(ODIM / BN, BDIM / BM), 256, 0, stream>>>(hbf, wt, bias0, out);
}
